// RifatAttention_36850819400541
// MI455X (gfx1250) — matrix operand order repaired and checked by reading, not compiled, not run on hardware
//
#include <hip/hip_runtime.h>
#include <hip/hip_bf16.h>

// ---------------- problem constants ----------------
#define GG   32
#define NHH  16
#define DD   1024
#define BBATCH 16
#define SSEQ 1025          // 32*32 + 1 (CLS)
#define SPAD 1056          // S padded to multiple of 32 (33*32)
#define MTOT (BBATCH * SSEQ)  // 16400 flattened rows for projections

typedef __attribute__((ext_vector_type(16))) __bf16 bf16x16;
typedef __attribute__((ext_vector_type(8)))  float  floatx8;

struct FragU { unsigned int u[8]; };

__device__ __forceinline__ unsigned short f32_to_bf16_rne(float f) {
    unsigned int u = __builtin_bit_cast(unsigned int, f);
    u += 0x7FFFu + ((u >> 16) & 1u);
    return (unsigned short)(u >> 16);
}

#define WMMA_BF16(A, B, C) \
    __builtin_amdgcn_wmma_f32_16x16x32_bf16(false, (A), false, (B), (short)0, (C), false, false)

// ---------------- fragment loaders (wave32, ISA 7.12.2 layouts) ----------------
// A: 16x32 bf16. lanes 0-15 rows 0-15 / low-K pattern, lanes 16-31 same rows / high-K pattern.
__device__ __forceinline__ bf16x16 load_a_frag(const unsigned short* __restrict__ base,
                                               int ld, int row_limit, int m0, int k0) {
    int lane = threadIdx.x & 31;
    int row  = lane & 15;
    int half = lane >> 4;
    int r = m0 + row; if (r >= row_limit) r = row_limit - 1;   // clamp; stores are predicated
    const unsigned short* p = base + (size_t)r * ld + k0;
    FragU f;
#pragma unroll
    for (int v = 0; v < 8; ++v) {
        int k = (v < 4) ? (half * 8 + v * 2) : (16 + half * 8 + (v - 4) * 2);
        f.u[v] = *(const unsigned int*)(p + k);
    }
    return __builtin_bit_cast(bf16x16, f);
}

// B: 32x16 bf16, B[kk,col]; source stored row-major as src[n][k] (contiguous K-pairs).
__device__ __forceinline__ bf16x16 load_b_frag(const unsigned short* __restrict__ base,
                                               int ld, int n_limit, int n0, int k0) {
    int lane = threadIdx.x & 31;
    int col  = lane & 15;
    int half = lane >> 4;
    int n = n0 + col; if (n >= n_limit) n = n_limit - 1;        // clamp; stores are predicated
    const unsigned short* p = base + (size_t)n * ld + k0;
    FragU f;
#pragma unroll
    for (int v = 0; v < 8; ++v) {
        int k = (v < 4) ? (half * 8 + v * 2) : (16 + half * 8 + (v - 4) * 2);
        f.u[v] = *(const unsigned int*)(p + k);
    }
    return __builtin_bit_cast(bf16x16, f);
}

// ---------------- utility kernels ----------------
__global__ void cvt_f32_to_bf16_kernel(const float* __restrict__ src,
                                       unsigned short* __restrict__ dst, int n) {
    int i = blockIdx.x * blockDim.x + threadIdx.x;
    int stride = gridDim.x * blockDim.x;
    for (; i < n; i += stride) dst[i] = f32_to_bf16_rne(src[i]);
}

__global__ void zero_u32_kernel(unsigned int* __restrict__ p, long long n) {
    long long i = (long long)blockIdx.x * blockDim.x + threadIdx.x;
    long long stride = (long long)gridDim.x * blockDim.x;
    for (; i < n; i += stride) p[i] = 0u;
}

// ============================================================================
// Pipelined 32x64-per-wave WMMA GEMM core:
//   acc[h*4+t] += A(rows m0+h*16 .. +15) x B(cols n0+t*16 .. +15)
// Fragments for step k+32 are issued BEFORE the 8 WMMAs of step k, so loads
// stay in flight across the matrix ops (waits become loadcnt<=N, not 0).
// Requires __launch_bounds__(256) on the calling kernel so ~180 VGPRs fit
// without spilling (8 waves per WGP).
// ============================================================================
#define GEMM_PIPELINE(Abase, Ald, Alim, Bbase, Bld, Blim, KDIM)                         \
    floatx8 acc[8] = {};                                                                \
    bf16x16 a0 = load_a_frag(Abase, Ald, Alim, m0,      0);                             \
    bf16x16 a1 = load_a_frag(Abase, Ald, Alim, m0 + 16, 0);                             \
    bf16x16 b0 = load_b_frag(Bbase, Bld, Blim, n0,      0);                             \
    bf16x16 b1 = load_b_frag(Bbase, Bld, Blim, n0 + 16, 0);                             \
    bf16x16 b2 = load_b_frag(Bbase, Bld, Blim, n0 + 32, 0);                             \
    bf16x16 b3 = load_b_frag(Bbase, Bld, Blim, n0 + 48, 0);                             \
    for (int k0 = 32; k0 < (KDIM); k0 += 32) {                                          \
        bf16x16 na0 = load_a_frag(Abase, Ald, Alim, m0,      k0);                       \
        bf16x16 na1 = load_a_frag(Abase, Ald, Alim, m0 + 16, k0);                       \
        bf16x16 nb0 = load_b_frag(Bbase, Bld, Blim, n0,      k0);                       \
        bf16x16 nb1 = load_b_frag(Bbase, Bld, Blim, n0 + 16, k0);                       \
        bf16x16 nb2 = load_b_frag(Bbase, Bld, Blim, n0 + 32, k0);                       \
        bf16x16 nb3 = load_b_frag(Bbase, Bld, Blim, n0 + 48, k0);                       \
        acc[0] = WMMA_BF16(a0, b0, acc[0]);                                             \
        acc[1] = WMMA_BF16(a0, b1, acc[1]);                                             \
        acc[2] = WMMA_BF16(a0, b2, acc[2]);                                             \
        acc[3] = WMMA_BF16(a0, b3, acc[3]);                                             \
        acc[4] = WMMA_BF16(a1, b0, acc[4]);                                             \
        acc[5] = WMMA_BF16(a1, b1, acc[5]);                                             \
        acc[6] = WMMA_BF16(a1, b2, acc[6]);                                             \
        acc[7] = WMMA_BF16(a1, b3, acc[7]);                                             \
        a0 = na0; a1 = na1; b0 = nb0; b1 = nb1; b2 = nb2; b3 = nb3;                     \
    }                                                                                   \
    acc[0] = WMMA_BF16(a0, b0, acc[0]);                                                 \
    acc[1] = WMMA_BF16(a0, b1, acc[1]);                                                 \
    acc[2] = WMMA_BF16(a0, b2, acc[2]);                                                 \
    acc[3] = WMMA_BF16(a0, b3, acc[3]);                                                 \
    acc[4] = WMMA_BF16(a1, b0, acc[4]);                                                 \
    acc[5] = WMMA_BF16(a1, b1, acc[5]);                                                 \
    acc[6] = WMMA_BF16(a1, b2, acc[6]);                                                 \
    acc[7] = WMMA_BF16(a1, b3, acc[7]);

// ---------------- projection GEMM: out[m,n] = sum_k A[m,k]*W[n,k] ----------------
// mode 0: dst[m*DD + n] = bf16(acc)           (q, k)
// mode 1: dst[(b*DD + n)*SPAD + s]            (v stored transposed, S-padded)
__global__ void __launch_bounds__(256)
proj_wmma_kernel(const unsigned short* __restrict__ A,
                 const unsigned short* __restrict__ W,
                 unsigned short* __restrict__ dst, int transpose_v) {
    const int mtiles = (MTOT + 31) / 32;    // 513
    const int nstrips = DD / 64;            // 16
    int wave = blockIdx.x * (blockDim.x >> 5) + (threadIdx.x >> 5);
    if (wave >= mtiles * nstrips) return;   // wave-uniform: EXEC stays all-ones for WMMA
    int mt = wave / nstrips, ns = wave % nstrips;
    int m0 = mt * 32, n0 = ns * 64;

    GEMM_PIPELINE(A, DD, MTOT, W, DD, DD, DD)

    int lane = threadIdx.x & 31;
    int col = lane & 15, half = lane >> 4;
#pragma unroll
    for (int h = 0; h < 2; ++h) {
#pragma unroll
        for (int t = 0; t < 4; ++t) {
            int n = n0 + t * 16 + col;
#pragma unroll
            for (int r = 0; r < 8; ++r) {
                int m = m0 + h * 16 + r + half * 8;
                if (m < MTOT) {
                    unsigned short hv = f32_to_bf16_rne(acc[h * 4 + t][r]);
                    if (!transpose_v) {
                        dst[(size_t)m * DD + n] = hv;
                    } else {
                        int b = m / SSEQ, s = m % SSEQ;
                        dst[((size_t)b * DD + n) * SPAD + s] = hv;
                    }
                }
            }
        }
    }
}

// ---------------- scores = q·k^T (+ relative-position bias), f32 out ----------------
__device__ __forceinline__ float rel_bias(const float* __restrict__ bt, int b, int q, int kc) {
    if (q < 1 || kc < 1) return 0.0f;
    int qq = q - 1, kk = kc - 1;
    int qi = qq >> 5, qj = qq & 31, ki = kk >> 5, kj = kk & 31;
    int idx = (qi - ki + GG - 1) * (2 * GG - 1) + (qj - kj + GG - 1);
    return bt[idx * NHH + b];
}

__global__ void __launch_bounds__(256)
scores_wmma_kernel(const unsigned short* __restrict__ Q,
                   const unsigned short* __restrict__ K,
                   const float* __restrict__ bias_table,
                   float* __restrict__ scores) {
    const int mtiles = (SSEQ + 31) / 32;    // 33
    const int nstrips = (SSEQ + 63) / 64;   // 17
    const int per_b = mtiles * nstrips;
    int wave = blockIdx.x * (blockDim.x >> 5) + (threadIdx.x >> 5);
    if (wave >= BBATCH * per_b) return;
    int b = wave / per_b;
    int rem = wave % per_b;
    int mt = rem / nstrips, ns = rem % nstrips;
    int m0 = mt * 32, n0 = ns * 64;

    const unsigned short* Qb = Q + (size_t)b * SSEQ * DD;
    const unsigned short* Kb = K + (size_t)b * SSEQ * DD;

    GEMM_PIPELINE(Qb, DD, SSEQ, Kb, DD, SSEQ, DD)

    int lane = threadIdx.x & 31;
    int col = lane & 15, half = lane >> 4;
#pragma unroll
    for (int h = 0; h < 2; ++h) {
#pragma unroll
        for (int t = 0; t < 4; ++t) {
            int n = n0 + t * 16 + col;
#pragma unroll
            for (int r = 0; r < 8; ++r) {
                int q = m0 + h * 16 + r + half * 8;
                if (q < SSEQ && n < SSEQ) {
                    scores[((size_t)b * SSEQ + q) * SSEQ + n] =
                        acc[h * 4 + t][r] + rel_bias(bias_table, b, q, n);
                }
            }
        }
    }
}

// ---------------- block-per-row softmax, emits zero-padded bf16 probs ----------------
__global__ void __launch_bounds__(256)
softmax_kernel(const float* __restrict__ scores,
               unsigned short* __restrict__ probs) {
    int row = blockIdx.x;                              // 0 .. B*S-1
    const float* srow = scores + (size_t)row * SSEQ;
    unsigned short* prow = probs + (size_t)row * SPAD;
    __shared__ float red[256];
    int tid = threadIdx.x;

    float m = -3.4e38f;
    for (int i = tid; i < SSEQ; i += 256) m = fmaxf(m, srow[i]);
    red[tid] = m; __syncthreads();
    for (int s = 128; s > 0; s >>= 1) {
        if (tid < s) red[tid] = fmaxf(red[tid], red[tid + s]);
        __syncthreads();
    }
    float rmax = red[0]; __syncthreads();

    float sum = 0.0f;
    for (int i = tid; i < SSEQ; i += 256) sum += __expf(srow[i] - rmax);
    red[tid] = sum; __syncthreads();
    for (int s = 128; s > 0; s >>= 1) {
        if (tid < s) red[tid] += red[tid + s];
        __syncthreads();
    }
    float inv = 1.0f / red[0];

    for (int i = tid; i < SPAD; i += 256) {
        float p = (i < SSEQ) ? __expf(srow[i] - rmax) * inv : 0.0f;
        prow[i] = f32_to_bf16_rne(p);
    }
}

// ---------------- out = probs · v  (uses vT so B-frags are contiguous) ----------------
__global__ void __launch_bounds__(256)
out_wmma_kernel(const unsigned short* __restrict__ P,
                const unsigned short* __restrict__ VT,
                float* __restrict__ out) {
    const int mtiles = (SSEQ + 31) / 32;    // 33
    const int nstrips = DD / 64;            // 16
    const int per_b = mtiles * nstrips;
    int wave = blockIdx.x * (blockDim.x >> 5) + (threadIdx.x >> 5);
    if (wave >= BBATCH * per_b) return;
    int b = wave / per_b;
    int rem = wave % per_b;
    int mt = rem / nstrips, ns = rem % nstrips;
    int m0 = mt * 32, n0 = ns * 64;

    const unsigned short* Pb = P  + (size_t)b * SSEQ * SPAD;
    const unsigned short* Vb = VT + (size_t)b * DD   * SPAD;

    GEMM_PIPELINE(Pb, SPAD, SSEQ, Vb, SPAD, DD, SPAD)   // pads are zero -> no K-edge handling

    int lane = threadIdx.x & 31;
    int col = lane & 15, half = lane >> 4;
#pragma unroll
    for (int h = 0; h < 2; ++h) {
#pragma unroll
        for (int t = 0; t < 4; ++t) {
            int n = n0 + t * 16 + col;
#pragma unroll
            for (int r = 0; r < 8; ++r) {
                int q = m0 + h * 16 + r + half * 8;
                if (q < SSEQ) out[((size_t)b * SSEQ + q) * DD + n] = acc[h * 4 + t][r];
            }
        }
    }
}

// ---------------- host-side orchestration ----------------
extern "C" void kernel_launch(void* const* d_in, const int* in_sizes, int n_in,
                              void* d_out, int out_size, void* d_ws, size_t ws_size,
                              hipStream_t stream) {
    const float* hidden = (const float*)d_in[0];   // [B,S,D]
    const float* Wq     = (const float*)d_in[1];   // [D,D]
    const float* Wk     = (const float*)d_in[2];
    const float* Wv     = (const float*)d_in[3];
    const float* btab   = (const float*)d_in[4];   // [(2G-1)^2, NH]
    float* out = (float*)d_out;

    // workspace carving (256B aligned)
    size_t off = 0;
    auto take = [&](size_t bytes) -> char* {
        char* p = (char*)d_ws + off;
        off = (off + bytes + 255) & ~(size_t)255;
        return p;
    };
    const size_t HB  = (size_t)MTOT * DD * 2;              // bf16 activations
    const size_t WB  = (size_t)DD * DD * 2;                // bf16 weights
    const size_t VTB = (size_t)BBATCH * DD * SPAD * 2;     // bf16 v-transposed (padded)
    const size_t SCB = (size_t)BBATCH * SSEQ * SSEQ * 4;   // f32 scores
    const size_t PRB = (size_t)BBATCH * SSEQ * SPAD * 2;   // bf16 probs (padded)

    unsigned short* h_bf  = (unsigned short*)take(HB);
    unsigned short* wq_bf = (unsigned short*)take(WB);
    unsigned short* wk_bf = (unsigned short*)take(WB);
    unsigned short* wv_bf = (unsigned short*)take(WB);
    unsigned short* q_bf  = (unsigned short*)take(HB);
    unsigned short* k_bf  = (unsigned short*)take(HB);
    unsigned short* vT_bf = (unsigned short*)take(VTB);
    float*          scor  = (float*)take(SCB);
    unsigned short* probs = (unsigned short*)take(PRB);

    // 1) convert inputs to bf16
    cvt_f32_to_bf16_kernel<<<2048, 256, 0, stream>>>(hidden, h_bf, MTOT * DD);
    cvt_f32_to_bf16_kernel<<<1024, 256, 0, stream>>>(Wq, wq_bf, DD * DD);
    cvt_f32_to_bf16_kernel<<<1024, 256, 0, stream>>>(Wk, wk_bf, DD * DD);
    cvt_f32_to_bf16_kernel<<<1024, 256, 0, stream>>>(Wv, wv_bf, DD * DD);

    // 2) zero vT (its S-padding must be zero for the probs·v GEMM)
    zero_u32_kernel<<<2048, 256, 0, stream>>>((unsigned int*)vT_bf, (long long)(VTB / 4));

    // 3) projections: q, k (row-major bf16) and v (transposed bf16)
    {
        int waves = ((MTOT + 31) / 32) * (DD / 64);      // 8208
        int blocks = (waves + 7) / 8;
        proj_wmma_kernel<<<blocks, 256, 0, stream>>>(h_bf, wq_bf, q_bf, 0);
        proj_wmma_kernel<<<blocks, 256, 0, stream>>>(h_bf, wk_bf, k_bf, 0);
        proj_wmma_kernel<<<blocks, 256, 0, stream>>>(h_bf, wv_bf, vT_bf, 1);
    }

    // 4) scores = q·k^T + bias
    {
        int waves = BBATCH * ((SSEQ + 31) / 32) * ((SSEQ + 63) / 64);  // 8976
        int blocks = (waves + 7) / 8;
        scores_wmma_kernel<<<blocks, 256, 0, stream>>>(q_bf, k_bf, btab, scor);
    }

    // 5) softmax -> padded bf16 probs
    softmax_kernel<<<BBATCH * SSEQ, 256, 0, stream>>>(scor, probs);

    // 6) out = probs · v
    {
        int waves = BBATCH * ((SSEQ + 31) / 32) * (DD / 64);           // 8448
        int blocks = (waves + 7) / 8;
        out_wmma_kernel<<<blocks, 256, 0, stream>>>(probs, vT_bf, out);
    }
    (void)in_sizes; (void)n_in; (void)out_size; (void)ws_size;
}
